// RNNModel_15762529976972
// MI455X (gfx1250) — matrix operand order repaired and checked by reading, not compiled, not run on hardware
//
#include <hip/hip_runtime.h>
#include <hip/hip_bf16.h>
#include <stdint.h>

// ---------------------------------------------------------------------------
// BRC RNN (2 layers) for MI455X / gfx1250.
//   - x and all weights pre-converted to bf16 once; every GEMM operand is
//     bf16 in global memory -> staging loops are pure copy (no VALU cvt),
//     half the read bytes of the f32 path.
//   - GEMMs: v_wmma_f32_16x16x32_bf16, 128x128 block tile, 32x64 wave tile,
//     double-buffered LDS (one barrier per k-tile) + global_prefetch of the
//     tile two steps ahead.
//   - BRC recurrence is diagonal -> scan is elementwise per (b,h) element,
//     one thread each, no synchronization.
// ---------------------------------------------------------------------------

#define S_LEN   512
#define BATCH   64
#define IN_DIM  512
#define H_DIM   1024
#define OUT_DIM 512

typedef __attribute__((ext_vector_type(16))) __bf16        v16bf;
typedef __attribute__((ext_vector_type(8)))  float         v8f;
typedef __attribute__((ext_vector_type(4)))  unsigned int  uint4v;

__device__ __forceinline__ unsigned short f32_to_bf16(float f) {
    unsigned int u = __float_as_uint(f);
    unsigned int r = u + 0x7FFFu + ((u >> 16) & 1u);   // round-to-nearest-even
    return (unsigned short)(r >> 16);
}

union Frag16 { uint4v u[2]; v16bf v; };

// C[M,N] = A[M,K] @ B[K,N] + bias[N].  A,B bf16; bias,C f32.
// 256 threads = 8 waves; block tile 128x128; wave tile 32x64 (2x4 WMMAs).
__global__ __launch_bounds__(256)
void gemm_wmma_bf16(const unsigned short* __restrict__ A,
                    const unsigned short* __restrict__ B,
                    const float* __restrict__ bias, float* __restrict__ C,
                    int M, int N, int K)
{
    __shared__ __align__(16) unsigned short As[2][128][40];  // [m][k], padded
    __shared__ __align__(16) unsigned short Bs[2][128][40];  // [n][k] transposed

    const int tid  = threadIdx.x;
    const int lane = tid & 31;
    const int wave = tid >> 5;      // 0..7
    const int wm   = wave & 3;      // 4 waves along M
    const int wn   = wave >> 2;     // 2 waves along N
    const int row0 = blockIdx.x * 128;
    const int col0 = blockIdx.y * 128;
    const int KT   = K >> 5;        // k-tiles of 32

    v8f acc[2][4] = {};

    const int half = lane >> 4;
    const int l15  = lane & 15;

    // stage one 128x32 A tile + 32x128 B tile (both bf16) into LDS buffer `buf`
    auto load_tile = [&](int kt, int buf) {
        const int k0 = kt << 5;
        // A: 2048 dwords (2 bf16 each), 8 per thread, coalesced 64B rows
        for (int idx = tid; idx < 128 * 16; idx += 256) {
            int r = idx >> 4, c2 = (idx & 15) << 1;
            unsigned int v = *(const unsigned int*)&A[(size_t)(row0 + r) * K + k0 + c2];
            *(unsigned int*)&As[buf][r][c2] = v;
        }
        // B: 32 rows x 128 cols; load dwords (2 cols), scatter transposed
        for (int idx = tid; idx < 32 * 64; idx += 256) {
            int kk = idx >> 6, n2 = (idx & 63) << 1;
            unsigned int v = *(const unsigned int*)&B[(size_t)(k0 + kk) * N + col0 + n2];
            Bs[buf][n2 + 0][kk] = (unsigned short)(v & 0xFFFFu);
            Bs[buf][n2 + 1][kk] = (unsigned short)(v >> 16);
        }
    };

    load_tile(0, 0);

    for (int kt = 0; kt < KT; ++kt) {
        __syncthreads();
        // prefetch tile kt+2 into caches while kt computes and kt+1 loads
        if (kt + 2 < KT) {
            const int kp = (kt + 2) << 5;
            if (tid < 128)
                __builtin_prefetch(&A[(size_t)(row0 + tid) * K + kp], 0, 0);
            else if (tid < 160)
                __builtin_prefetch(&B[(size_t)(kp + (tid - 128)) * N + col0], 0, 0);
            else if (tid < 192)
                __builtin_prefetch(&B[(size_t)(kp + (tid - 160)) * N + col0 + 64], 0, 0);
        }
        if (kt + 1 < KT) load_tile(kt + 1, (kt + 1) & 1);

        const int buf = kt & 1;
        // per-lane fragments per ISA 7.12.2 layouts
        Frag16 fa[2], fb[4];
        #pragma unroll
        for (int i = 0; i < 2; ++i) {
            int m = wm * 32 + i * 16 + l15;
            fa[i].u[0] = *(const uint4v*)&As[buf][m][half * 8];       // K=h*8..+8
            fa[i].u[1] = *(const uint4v*)&As[buf][m][16 + half * 8];  // K=16+h*8..+8
        }
        #pragma unroll
        for (int j = 0; j < 4; ++j) {
            int n = wn * 64 + j * 16 + l15;
            fb[j].u[0] = *(const uint4v*)&Bs[buf][n][half * 8];       // K=h*8..+8
            fb[j].u[1] = *(const uint4v*)&Bs[buf][n][16 + half * 8];  // K=16+h*8..+8
        }
        #pragma unroll
        for (int i = 0; i < 2; ++i)
            #pragma unroll
            for (int j = 0; j < 4; ++j)
                acc[i][j] = __builtin_amdgcn_wmma_f32_16x16x32_bf16(
                    false, fa[i].v, false, fb[j].v,
                    (short)0, acc[i][j], false, false);
    }

    // epilogue: C/D layout (row = r + half*8, col = lane&15) -> global, + bias
    const int r8 = half * 8;
    #pragma unroll
    for (int i = 0; i < 2; ++i) {
        int rowb = row0 + wm * 32 + i * 16 + r8;
        #pragma unroll
        for (int j = 0; j < 4; ++j) {
            int col = col0 + wn * 64 + j * 16 + l15;
            float bv = bias[col];
            #pragma unroll
            for (int r = 0; r < 8; ++r)
                C[(size_t)(rowb + r) * N + col] = acc[i][j][r] + bv;
        }
    }
}

// f32 -> bf16 one-shot conversion (x and weights), vectorized 4-wide.
__global__ __launch_bounds__(256)
void cvt_bf16_kernel(const float* __restrict__ src, unsigned short* __restrict__ dst,
                     int n4)
{
    int i = blockIdx.x * blockDim.x + threadIdx.x;
    if (i >= n4) return;
    const float4 f = ((const float4*)src)[i];
    unsigned int lo = (unsigned int)f32_to_bf16(f.x) | ((unsigned int)f32_to_bf16(f.y) << 16);
    unsigned int hi = (unsigned int)f32_to_bf16(f.z) | ((unsigned int)f32_to_bf16(f.w) << 16);
    ((unsigned int*)dst)[2 * i + 0] = lo;
    ((unsigned int*)dst)[2 * i + 1] = hi;
}

// Diagonal BRC recurrence: one thread per (b,h) element, loop over time.
__global__ __launch_bounds__(256)
void brc_scan(const float* __restrict__ xa, const float* __restrict__ xc,
              const float* __restrict__ xh,
              const float* __restrict__ wa, const float* __restrict__ wc,
              const float* __restrict__ h_init,
              unsigned short* __restrict__ y_bf16,  // [S,B,H] bf16
              float* __restrict__ h_last,           // [B,H]
              int Sn, int BH, int H)
{
    int i = blockIdx.x * blockDim.x + threadIdx.x;
    if (i >= BH) return;
    const int j   = i & (H - 1);
    const float waj = wa[j];
    const float wcj = wc[j];
    float h = h_init[i];
    size_t off = (size_t)i;
    for (int t = 0; t < Sn; ++t, off += (size_t)BH) {
        float a  = 1.0f + tanhf(xa[off] + waj * h);
        float cg = xc[off] + wcj * h;
        float c  = 1.0f / (1.0f + __expf(-cg));
        h = c * h + (1.0f - c) * tanhf(xh[off] + a * h);
        y_bf16[off] = f32_to_bf16(h);
    }
    h_last[i] = h;
}

extern "C" void kernel_launch(void* const* d_in, const int* in_sizes, int n_in,
                              void* d_out, int out_size, void* d_ws, size_t ws_size,
                              hipStream_t stream)
{
    const float* x    = (const float*)d_in[0];
    const float* h0   = (const float*)d_in[1];   // [L,B,H]
    const float* Ua0  = (const float*)d_in[2];
    const float* Uc0  = (const float*)d_in[3];
    const float* Uh0  = (const float*)d_in[4];
    const float* wa0  = (const float*)d_in[5];
    const float* wc0  = (const float*)d_in[6];
    const float* ba0  = (const float*)d_in[7];
    const float* bc0  = (const float*)d_in[8];
    const float* bh0  = (const float*)d_in[9];
    const float* Ua1  = (const float*)d_in[10];
    const float* Uc1  = (const float*)d_in[11];
    const float* Uh1  = (const float*)d_in[12];
    const float* wa1  = (const float*)d_in[13];
    const float* wc1  = (const float*)d_in[14];
    const float* ba1  = (const float*)d_in[15];
    const float* bc1  = (const float*)d_in[16];
    const float* bh1  = (const float*)d_in[17];
    const float* Wdec = (const float*)d_in[18];
    const float* bdec = (const float*)d_in[19];

    float* out = (float*)d_out;
    const int M  = S_LEN * BATCH;          // 32768
    const int BH = BATCH * H_DIM;          // 65536
    const size_t MH = (size_t)M * H_DIM;

    // workspace layout
    char* ws = (char*)d_ws;
    float* xa = (float*)ws;                    ws += MH * 4;
    float* xc = (float*)ws;                    ws += MH * 4;
    float* xh = (float*)ws;                    ws += MH * 4;
    unsigned short* y0  = (unsigned short*)ws; ws += MH * 2;
    unsigned short* y1  = (unsigned short*)ws; ws += MH * 2;
    unsigned short* xb  = (unsigned short*)ws; ws += (size_t)M * IN_DIM * 2;
    unsigned short* ua0 = (unsigned short*)ws; ws += (size_t)IN_DIM * H_DIM * 2;
    unsigned short* uc0 = (unsigned short*)ws; ws += (size_t)IN_DIM * H_DIM * 2;
    unsigned short* uh0 = (unsigned short*)ws; ws += (size_t)IN_DIM * H_DIM * 2;
    unsigned short* ua1 = (unsigned short*)ws; ws += (size_t)H_DIM * H_DIM * 2;
    unsigned short* uc1 = (unsigned short*)ws; ws += (size_t)H_DIM * H_DIM * 2;
    unsigned short* uh1 = (unsigned short*)ws; ws += (size_t)H_DIM * H_DIM * 2;
    unsigned short* wdc = (unsigned short*)ws; ws += (size_t)H_DIM * OUT_DIM * 2;

    float* hid = out + (size_t)M * OUT_DIM;    // hidden tail [L,B,H]

    dim3 blk(256);
    auto cvt = [&](const float* s, unsigned short* d, size_t n) {
        int n4 = (int)(n >> 2);
        cvt_bf16_kernel<<<(n4 + 255) / 256, blk, 0, stream>>>(s, d, n4);
    };

    // one-shot bf16 conversions
    cvt(x,    xb,  (size_t)M * IN_DIM);
    cvt(Ua0,  ua0, (size_t)IN_DIM * H_DIM);
    cvt(Uc0,  uc0, (size_t)IN_DIM * H_DIM);
    cvt(Uh0,  uh0, (size_t)IN_DIM * H_DIM);
    cvt(Ua1,  ua1, (size_t)H_DIM * H_DIM);
    cvt(Uc1,  uc1, (size_t)H_DIM * H_DIM);
    cvt(Uh1,  uh1, (size_t)H_DIM * H_DIM);
    cvt(Wdec, wdc, (size_t)H_DIM * OUT_DIM);

    dim3 gH(M / 128, H_DIM / 128);
    dim3 gO(M / 128, OUT_DIM / 128);
    dim3 gScan(BH / 256);

    // layer 0: input->hidden projections, then scan
    gemm_wmma_bf16<<<gH, blk, 0, stream>>>(xb, ua0, ba0, xa, M, H_DIM, IN_DIM);
    gemm_wmma_bf16<<<gH, blk, 0, stream>>>(xb, uc0, bc0, xc, M, H_DIM, IN_DIM);
    gemm_wmma_bf16<<<gH, blk, 0, stream>>>(xb, uh0, bh0, xh, M, H_DIM, IN_DIM);
    brc_scan<<<gScan, blk, 0, stream>>>(xa, xc, xh, wa0, wc0, h0, y0, hid,
                                        S_LEN, BH, H_DIM);

    // layer 1: hidden->hidden projections (A = y0 bf16), then scan
    gemm_wmma_bf16<<<gH, blk, 0, stream>>>(y0, ua1, ba1, xa, M, H_DIM, H_DIM);
    gemm_wmma_bf16<<<gH, blk, 0, stream>>>(y0, uc1, bc1, xc, M, H_DIM, H_DIM);
    gemm_wmma_bf16<<<gH, blk, 0, stream>>>(y0, uh1, bh1, xh, M, H_DIM, H_DIM);
    brc_scan<<<gScan, blk, 0, stream>>>(xa, xc, xh, wa1, wc1, h0 + BH, y1, hid + BH,
                                        S_LEN, BH, H_DIM);

    // decoder: out = y1 @ Wdec + bdec
    gemm_wmma_bf16<<<gO, blk, 0, stream>>>(y1, wdc, bdec, out,
                                           M, OUT_DIM, H_DIM);
}
